// MinimalChiralModel_11982958756599
// MI455X (gfx1250) — matrix operand order repaired and checked by reading, not compiled, not run on hardware
//
#include <hip/hip_runtime.h>
#include <hip/hip_bf16.h>
#include <math.h>

// ---------------------------------------------------------------------------
// MinimalChiralModel on MI455X (gfx1250), wave32 + WMMA f32_16x16x32_f16.
// Memory-bound (~0.8 GB traffic, ~2.6 GFLOP): D=64 GEMMs on WMMA with
// f16 operands / f32 accumulate; A tiles staged to LDS via coalesced float4
// loads; weights staged transposed in LDS so B fragments are contiguous
// 32-byte ds reads. Weight staging for the hinge/recon GEMMs goes through the
// Tensor Data Mover (tensor_load_to_lds + s_wait_tensorcnt) when available.
// ---------------------------------------------------------------------------

#define N_NODES 50000
#define N_EDGES 800000
#define POOL_K  25000
#define NBLK_N  196          // ceil(N_NODES/256)
#define NBLK_K  196          // ceil(POOL_K/128)
#define CYC_BLK 512

typedef __attribute__((ext_vector_type(16))) _Float16 v16h;
typedef __attribute__((ext_vector_type(8)))  _Float16 v8h;
typedef __attribute__((ext_vector_type(8)))  float    v8f;
typedef __attribute__((ext_vector_type(4)))  unsigned u32x4;
typedef __attribute__((ext_vector_type(8)))  int      i32x8;
typedef __attribute__((ext_vector_type(4)))  int      i32x4;

// ---- workspace layout (4-byte units) --------------------------------------
#define OFF_XB     ((size_t)0)                         // [N,2,64] x@basis
#define OFF_XL1    (OFF_XB   + (size_t)N_NODES*128)    // [N,64] x@w_root -> x_l1
#define OFF_AGG    (OFF_XL1  + (size_t)N_NODES*64)     // [N,64]
#define OFF_DEG    (OFF_AGG  + (size_t)N_NODES*64)     // [N]
#define OFF_HIST   (OFF_DEG  + (size_t)N_NODES)        // [256] u32
#define OFF_STATE  (OFF_HIST + 256)                    // [8]   u32 (prefix,kneed,gtot)
#define OFF_SCORE  (OFF_STATE+ 8)                      // [N]
#define OFF_KEYS   (OFF_SCORE+ (size_t)N_NODES)        // [N] u32
#define OFF_BLKG   (OFF_KEYS + (size_t)N_NODES)        // [NBLK_N] u32
#define OFF_BLKT   (OFF_BLKG + NBLK_N)
#define OFF_PREFG  (OFF_BLKT + NBLK_N)
#define OFF_PREFT  (OFF_PREFG+ NBLK_N)
#define OFF_PERMI  (OFF_PREFT+ NBLK_N)                 // [K] i32
#define OFF_ROWS   (OFF_PERMI+ (size_t)POOL_K)         // down[64],lower[64],asig[64],bsig[64],normp
#define OFF_TUL    (OFF_ROWS + 320)                    // [K,64] pre-LN hinge GEMM
#define OFF_CSP    (OFF_TUL  + (size_t)POOL_K*64)      // [NBLK_K,64] colsum partials
#define OFF_CS     (OFF_CSP  + (size_t)NBLK_K*64)      // [64]
#define OFF_CYC    (OFF_CS   + 64)                     // [CYC_BLK]

// ---- output layout (floats) -----------------------------------------------
#define O_LOGITS 0
#define O_UP     ((size_t)16)
#define O_DOWN   (O_UP   + (size_t)POOL_K*64)
#define O_CHIRAL (O_DOWN + (size_t)POOL_K*64)
#define O_REC    (O_CHIRAL + (size_t)POOL_K*64)
#define O_CYCLE  (O_REC  + (size_t)N_NODES*64)
#define O_PERM   (O_CYCLE + 1)
#define O_SCOREK (O_PERM + (size_t)POOL_K)

// ===========================================================================
// WMMA fragment helpers (ISA 7.12.2 wave32 layouts)
// ===========================================================================

// Stage one 16x64 f32 row tile (rows m0..m0+15, clamped) into LDS as f16.
// lane -> row (lane>>1), half-row (lane&1)*32 : fully coalesced float4 loads.
__device__ __forceinline__ void stage_a_tile(const float* __restrict__ X, int M,
                                             int m0, int lane,
                                             _Float16* __restrict__ tile /*16*64*/) {
    int r  = lane >> 1;
    int c0 = (lane & 1) * 32;
    int row = m0 + r; if (row >= M) row = M - 1;
    const float4* src = (const float4*)(X + (size_t)row * 64 + c0);
    _Float16* dst = tile + r * 64 + c0;
#pragma unroll
    for (int j = 0; j < 8; ++j) {
        float4 v = src[j];
        dst[j * 4 + 0] = (_Float16)v.x;
        dst[j * 4 + 1] = (_Float16)v.y;
        dst[j * 4 + 2] = (_Float16)v.z;
        dst[j * 4 + 3] = (_Float16)v.w;
    }
}

// A fragment (16x32) from LDS tile: two contiguous 8-half runs -> b128 loads.
__device__ __forceinline__ v16h frag_a(const _Float16* __restrict__ tile,
                                       int lane, int k0) {
    const int row = lane & 15;
    const int kbase = (lane < 16) ? 0 : 16;
    const _Float16* p = tile + row * 64 + k0 + kbase;
    v8h lo = *(const v8h*)p;          // K = k0+kbase+0..7
    v8h hi = *(const v8h*)(p + 8);    // K = k0+kbase+8..15
    return __builtin_shufflevector(lo, hi, 0, 1, 2, 3, 4, 5, 6, 7,
                                           8, 9, 10, 11, 12, 13, 14, 15);
}

// B fragment (32x16) from LDS weights stored TRANSPOSED (Wt[n*64+k]):
// 16 contiguous halves -> one aligned 32-byte LDS read.
__device__ __forceinline__ v16h frag_b(const _Float16* __restrict__ Wt,
                                       int lane, int n0, int k0) {
    const int col = n0 + (lane & 15);
    const int kb = (lane < 16) ? 0 : 16;
    return *(const v16h*)(Wt + col * 64 + k0 + kb);
}

// Issue a TDM load of a 64x64 f32 tensor (row-major) into LDS (thread 0 only),
// then wait on TENSORcnt. Caller must __syncthreads() after.
__device__ __forceinline__ void tdm_load_w64(const float* __restrict__ W,
                                             float* __restrict__ WldsF32) {
#if __has_builtin(__builtin_amdgcn_tensor_load_to_lds)
    if (threadIdx.x == 0) {
        // D# group0: count=1 | lds_addr | 57-bit global addr | type=2
        unsigned lds = (unsigned)(unsigned long long)(void*)WldsF32;
        unsigned long long ga = (unsigned long long)W;
        u32x4 g0;
        g0[0] = 1u;
        g0[1] = lds;
        g0[2] = (unsigned)ga;
        g0[3] = (unsigned)((ga >> 32) & 0x1FFFFFFull) | (2u << 30);
        // D# group1: data_size=4B, tensor 64x64, tile 64x64, dim0_stride=64
        i32x8 g1;
        g1[0] = (int)(2u << 16);          // data_size=2 (4 bytes), mask=0
        g1[1] = (int)(64u << 16);         // tensor_dim0 (bits 79:48) low half
        g1[2] = (int)(64u << 16);         // tensor_dim0 hi=0 | tensor_dim1 low
        g1[3] = (int)(64u << 16);         // tensor_dim1 hi=0 | tile_dim0=64
        g1[4] = 64;                       // tile_dim1=64, tile_dim2=0
        g1[5] = 64;                       // tensor_dim0_stride=64
        g1[6] = 0;
        g1[7] = 0;
        i32x4 z4 = {};
#if __clang_major__ >= 23
        i32x8 z8 = {};
        __builtin_amdgcn_tensor_load_to_lds(g0, g1, z4, z4, z8, 0);
#else
        __builtin_amdgcn_tensor_load_to_lds(g0, g1, z4, z4, 0);
#endif
        __builtin_amdgcn_s_wait_tensorcnt((short)0);
    }
#endif
}

// ===========================================================================
// Kernels
// ===========================================================================
__global__ void fill_zero_kernel(float* p, size_t n) {
    for (size_t i = (size_t)blockIdx.x * blockDim.x + threadIdx.x; i < n;
         i += (size_t)gridDim.x * blockDim.x) p[i] = 0.f;
}

// rows: down_row, lower_t row, sigmoid(alpha/beta), ||pool_p||  (single block)
__global__ void rows_kernel(const float* l3_prior, const float* w_unpool, const float* b_unpool,
                            const float* w_lu, const float* b_lu, const float* g_lu,
                            const float* bn_lu, const float* alpha, const float* beta,
                            const float* pool_p, float* rows) {
    __shared__ float dn[64], pre[64], red[64];
    int c = threadIdx.x;                       // 64 threads
    float v = b_unpool[c];
    for (int d = 0; d < 64; ++d) v += l3_prior[d] * w_unpool[d * 64 + c];
    dn[c] = v; rows[c] = v;
    red[c] = pool_p[c] * pool_p[c];
    __syncthreads();
    float p = b_lu[c];
    for (int d = 0; d < 64; ++d) p += dn[d] * w_lu[d * 64 + c];
    pre[c] = p;
    __syncthreads();
    if (c == 0) {
        float s = 0; for (int d = 0; d < 64; ++d) s += red[d];
        rows[256] = sqrtf(s);
        float mu = 0; for (int d = 0; d < 64; ++d) mu += pre[d];
        mu *= (1.f / 64.f);
        float var = 0; for (int d = 0; d < 64; ++d) { float x = pre[d] - mu; var += x * x; }
        var *= (1.f / 64.f);
        red[0] = mu; red[1] = rsqrtf(var + 1e-5f);
    }
    __syncthreads();
    float h = (pre[c] - red[0]) * red[1] * g_lu[c] + bn_lu[c];
    rows[64 + c]  = 0.5f * h * (1.f + erff(h * 0.7071067811865475f));
    rows[128 + c] = 1.f / (1.f + expf(-alpha[c]));
    rows[192 + c] = 1.f / (1.f + expf(-beta[c]));
}

// fused triple GEMM: xb0 = x@basis0, xb1 = x@basis1, xw = x@w_root  (WMMA)
__global__ void rgcn_gemms_kernel(const float* __restrict__ x, const float* __restrict__ basis,
                                  const float* __restrict__ w_root,
                                  float* __restrict__ xb, float* __restrict__ xw, int M) {
    __shared__ _Float16 Wt[3 * 4096];          // transposed weights, 24KB
    __shared__ _Float16 At[8][1024];           // per-wave A tiles, 16KB
    for (int i = threadIdx.x; i < 4096; i += blockDim.x) {
        int tr = (i & 63) * 64 + (i >> 6);
        Wt[tr]            = (_Float16)basis[i];
        Wt[4096 + tr]     = (_Float16)basis[4096 + i];
        Wt[2 * 4096 + tr] = (_Float16)w_root[i];
    }
    __syncthreads();
    const int lane = threadIdx.x & 31;
    const int wave = __builtin_amdgcn_readfirstlane(threadIdx.x >> 5);   // scalar
    const int m0 = (blockIdx.x * 8 + wave) * 16;                         // scalar
    if (m0 >= M) return;                       // scalar branch, after barrier
    stage_a_tile(x, M, m0, lane, At[wave]);
    v16h a0 = frag_a(At[wave], lane, 0);
    v16h a1 = frag_a(At[wave], lane, 32);
    const int col = lane & 15;
    const int rowoff = (lane < 16) ? 0 : 8;
    const bool full = (m0 + 16 <= M);          // scalar
#pragma unroll
    for (int w = 0; w < 3; ++w) {
        const _Float16* Wp = Wt + w * 4096;
#pragma unroll
        for (int t = 0; t < 4; ++t) {
            const int n0 = t * 16;
            v16h b0 = frag_b(Wp, lane, n0, 0);
            v16h b1 = frag_b(Wp, lane, n0, 32);
            v8f c = {};
            c = __builtin_amdgcn_wmma_f32_16x16x32_f16(false, a0, false, b0, (short)0, c, false, false);
            c = __builtin_amdgcn_wmma_f32_16x16x32_f16(false, a1, false, b1, (short)0, c, false, false);
            if (full) {
                if (w < 2) {
                    float* yp = xb + ((size_t)(m0 + rowoff) * 2 + w) * 64 + n0 + col;
#pragma unroll
                    for (int r = 0; r < 8; ++r) yp[(size_t)r * 128] = c[r];
                } else {
                    float* yp = xw + (size_t)(m0 + rowoff) * 64 + n0 + col;
#pragma unroll
                    for (int r = 0; r < 8; ++r) yp[(size_t)r * 64] = c[r];
                }
            } else {
#pragma unroll
                for (int r = 0; r < 8; ++r) {
                    int m = m0 + rowoff + r;
                    if (m < M) {
                        if (w < 2) xb[((size_t)m * 2 + w) * 64 + n0 + col] = c[r];
                        else       xw[(size_t)m * 64 + n0 + col] = c[r];
                    }
                }
            }
        }
    }
}

// generic Y = X@W + bias (M x 64 @ 64 x 64), WMMA; weights staged via TDM.
__global__ void gemm64_bias_kernel(const float* __restrict__ X, const float* __restrict__ W,
                                   const float* __restrict__ bias, float* __restrict__ Y, int M) {
    __shared__ _Float16 Wt[4096];
    __shared__ _Float16 At[8][1024];
#if __has_builtin(__builtin_amdgcn_tensor_load_to_lds)
    __shared__ float Wf32[4096];
    tdm_load_w64(W, Wf32);
    __syncthreads();
    for (int i = threadIdx.x; i < 4096; i += blockDim.x)
        Wt[(i & 63) * 64 + (i >> 6)] = (_Float16)Wf32[i];
#else
    for (int i = threadIdx.x; i < 4096; i += blockDim.x)
        Wt[(i & 63) * 64 + (i >> 6)] = (_Float16)W[i];
#endif
    __syncthreads();
    const int lane = threadIdx.x & 31;
    const int wave = __builtin_amdgcn_readfirstlane(threadIdx.x >> 5);
    const int m0 = (blockIdx.x * 8 + wave) * 16;
    if (m0 >= M) return;
    stage_a_tile(X, M, m0, lane, At[wave]);
    v16h a0 = frag_a(At[wave], lane, 0);
    v16h a1 = frag_a(At[wave], lane, 32);
    const int col = lane & 15;
    const int rowoff = (lane < 16) ? 0 : 8;
    const bool full = (m0 + 16 <= M);
#pragma unroll
    for (int t = 0; t < 4; ++t) {
        const int n0 = t * 16;
        v16h b0 = frag_b(Wt, lane, n0, 0);
        v16h b1 = frag_b(Wt, lane, n0, 32);
        v8f c = {};
        c = __builtin_amdgcn_wmma_f32_16x16x32_f16(false, a0, false, b0, (short)0, c, false, false);
        c = __builtin_amdgcn_wmma_f32_16x16x32_f16(false, a1, false, b1, (short)0, c, false, false);
        float bb = bias[n0 + col];
        if (full) {
            float* yp = Y + (size_t)(m0 + rowoff) * 64 + n0 + col;
#pragma unroll
            for (int r = 0; r < 8; ++r) yp[(size_t)r * 64] = c[r] + bb;
        } else {
#pragma unroll
            for (int r = 0; r < 8; ++r) {
                int m = m0 + rowoff + r;
                if (m < M) Y[(size_t)m * 64 + n0 + col] = c[r] + bb;
            }
        }
    }
}

// recon = chiral@w_rec + b_rec, scattered to x_l1_rec[perm[m]]  (WMMA + TDM)
__global__ void recon_scatter_kernel(const float* __restrict__ Xc, const float* __restrict__ W,
                                     const float* __restrict__ bias, const int* __restrict__ permI,
                                     float* __restrict__ Yrec, int M) {
    __shared__ _Float16 Wt[4096];
    __shared__ _Float16 At[8][1024];
#if __has_builtin(__builtin_amdgcn_tensor_load_to_lds)
    __shared__ float Wf32[4096];
    tdm_load_w64(W, Wf32);
    __syncthreads();
    for (int i = threadIdx.x; i < 4096; i += blockDim.x)
        Wt[(i & 63) * 64 + (i >> 6)] = (_Float16)Wf32[i];
#else
    for (int i = threadIdx.x; i < 4096; i += blockDim.x)
        Wt[(i & 63) * 64 + (i >> 6)] = (_Float16)W[i];
#endif
    __syncthreads();
    const int lane = threadIdx.x & 31;
    const int wave = __builtin_amdgcn_readfirstlane(threadIdx.x >> 5);
    const int m0 = (blockIdx.x * 8 + wave) * 16;
    if (m0 >= M) return;
    stage_a_tile(Xc, M, m0, lane, At[wave]);
    v16h a0 = frag_a(At[wave], lane, 0);
    v16h a1 = frag_a(At[wave], lane, 32);
    const int col = lane & 15;
    const int rowoff = (lane < 16) ? 0 : 8;
    const bool full = (m0 + 16 <= M);
#pragma unroll
    for (int t = 0; t < 4; ++t) {
        const int n0 = t * 16;
        v16h b0 = frag_b(Wt, lane, n0, 0);
        v16h b1 = frag_b(Wt, lane, n0, 32);
        v8f c = {};
        c = __builtin_amdgcn_wmma_f32_16x16x32_f16(false, a0, false, b0, (short)0, c, false, false);
        c = __builtin_amdgcn_wmma_f32_16x16x32_f16(false, a1, false, b1, (short)0, c, false, false);
        float bb = bias[n0 + col];
        if (full) {
#pragma unroll
            for (int r = 0; r < 8; ++r)
                Yrec[(size_t)permI[m0 + rowoff + r] * 64 + n0 + col] = c[r] + bb;
        } else {
#pragma unroll
            for (int r = 0; r < 8; ++r) {
                int m = m0 + rowoff + r;
                if (m < M) Yrec[(size_t)permI[m] * 64 + n0 + col] = c[r] + bb;
            }
        }
    }
}

// one wave per edge: msg = comp[et,0]*xb[src,0,:] + comp[et,1]*xb[src,1,:]; scatter-add
__global__ void edge_msg_kernel(const int* __restrict__ ei, const int* __restrict__ et,
                                const float* __restrict__ comp, const float* __restrict__ xb,
                                float* __restrict__ agg, float* __restrict__ deg, int E) {
    int e = (int)(((size_t)blockIdx.x * blockDim.x + threadIdx.x) >> 5);
    int lane = threadIdx.x & 31;
    if (e >= E) return;
    int s = ei[e], d = ei[E + e], ty = et[e];
    float c0 = comp[ty * 2 + 0], c1 = comp[ty * 2 + 1];
    const float2* xs0 = (const float2*)(xb + (size_t)s * 128);
    const float2* xs1 = (const float2*)(xb + (size_t)s * 128 + 64);
    float2 u = xs0[lane], w = xs1[lane];
    int c = lane * 2;
    atomicAdd(&agg[(size_t)d * 64 + c],     c0 * u.x + c1 * w.x);
    atomicAdd(&agg[(size_t)d * 64 + c + 1], c0 * u.y + c1 * w.y);
    if (lane == 0) atomicAdd(&deg[d], 1.0f);
}

// x_l1 = xw + agg/max(deg,1) + b_rgcn; score; radix keys (flip for descending)
__global__ void finalize_l1_kernel(float* __restrict__ x_l1, const float* __restrict__ agg,
                                   const float* __restrict__ deg, const float* __restrict__ b_rgcn,
                                   const float* __restrict__ pool_p, const float* __restrict__ rows,
                                   float* __restrict__ score, unsigned* __restrict__ keys, int N) {
    int n = blockIdx.x * blockDim.x + threadIdx.x;
    if (n >= N) return;
    float inv = 1.0f / fmaxf(deg[n], 1.0f);
    float s = 0.f;
    for (int c = 0; c < 64; ++c) {
        float v = x_l1[(size_t)n * 64 + c] + agg[(size_t)n * 64 + c] * inv + b_rgcn[c];
        x_l1[(size_t)n * 64 + c] = v;
        s += v * pool_p[c];
    }
    s /= rows[256];                            // / ||pool_p||
    score[n] = s;
    unsigned u = __float_as_uint(s);
    keys[n] = (u & 0x80000000u) ? ~u : (u | 0x80000000u);
}

__global__ void radix_hist_kernel(const unsigned* __restrict__ keys, int N,
                                  const unsigned* __restrict__ state, int pass,
                                  unsigned* __restrict__ hist) {
    int i = blockIdx.x * blockDim.x + threadIdx.x;
    if (i >= N) return;
    unsigned key = keys[i];
    int shift = 24 - 8 * pass;
    if (pass > 0 && (key >> (shift + 8)) != state[0]) return;
    atomicAdd(&hist[(key >> shift) & 255u], 1u);
}

__global__ void radix_pick_kernel(unsigned* hist, unsigned* state, int pass, int K) {
    if (threadIdx.x == 0) {
        unsigned kneed = (pass == 0) ? (unsigned)K : state[1];
        unsigned cum = 0, above = 0; int digit = 0;
        for (int b = 255; b >= 0; --b) {
            unsigned h = hist[b];
            if (cum + h >= kneed) { digit = b; above = cum; break; }
            cum += h;
        }
        state[0] = (pass == 0) ? (unsigned)digit : ((state[0] << 8) | (unsigned)digit);
        state[1] = kneed - above;                          // remaining within chosen bin
        state[2] = (pass == 0) ? above : state[2] + above; // total strictly-greater
    }
    __syncthreads();
    hist[threadIdx.x] = 0;                                 // ready for next pass
}

__global__ void select_flags_kernel(const unsigned* __restrict__ keys, int N,
                                    const unsigned* __restrict__ state,
                                    unsigned* blkG, unsigned* blkT) {
    __shared__ unsigned sg[256], st[256];
    int i = blockIdx.x * 256 + threadIdx.x;
    unsigned T = state[0], g = 0, t = 0;
    if (i < N) { unsigned k = keys[i]; g = (k > T); t = (k == T); }
    sg[threadIdx.x] = g; st[threadIdx.x] = t;
    __syncthreads();
    for (int s = 128; s > 0; s >>= 1) {
        if (threadIdx.x < s) { sg[threadIdx.x] += sg[threadIdx.x + s]; st[threadIdx.x] += st[threadIdx.x + s]; }
        __syncthreads();
    }
    if (threadIdx.x == 0) { blkG[blockIdx.x] = sg[0]; blkT[blockIdx.x] = st[0]; }
}

__global__ void block_scan_kernel(const unsigned* blkG, const unsigned* blkT,
                                  unsigned* prefG, unsigned* prefT, int nb) {
    if (threadIdx.x == 0) {
        unsigned a = 0, b = 0;
        for (int i = 0; i < nb; ++i) { prefG[i] = a; prefT[i] = b; a += blkG[i]; b += blkT[i]; }
    }
}

__global__ void emit_perm_kernel(const unsigned* __restrict__ keys, const float* __restrict__ score,
                                 int N, int K, const unsigned* __restrict__ state,
                                 const unsigned* __restrict__ prefG, const unsigned* __restrict__ prefT,
                                 int* __restrict__ permI, float* __restrict__ perm_f,
                                 float* __restrict__ score_k) {
    __shared__ unsigned sg[256], st[256];
    int tid = threadIdx.x, i = blockIdx.x * 256 + tid;
    unsigned T = state[0], ties_needed = state[1], gtot = state[2];
    unsigned g = 0, t = 0;
    if (i < N) { unsigned k = keys[i]; g = (k > T); t = (k == T); }
    sg[tid] = g; st[tid] = t;
    __syncthreads();
    for (int s = 1; s < 256; s <<= 1) {                    // Hillis-Steele inclusive
        unsigned vg = (tid >= s) ? sg[tid - s] : 0u;
        unsigned vt = (tid >= s) ? st[tid - s] : 0u;
        __syncthreads();
        sg[tid] += vg; st[tid] += vt;
        __syncthreads();
    }
    if (i < N) {
        unsigned exg = sg[tid] - g, ext = st[tid] - t;
        int pos = -1;
        if (g) pos = (int)(prefG[blockIdx.x] + exg);
        else if (t) {
            unsigned tr = prefT[blockIdx.x] + ext;
            if (tr < ties_needed) pos = (int)(gtot + tr);
        }
        if (pos >= 0 && pos < K) {
            permI[pos] = i;
            perm_f[pos] = (float)i;
            score_k[pos] = score[i];
        }
    }
}

__global__ void gather_up_kernel(const int* __restrict__ permI, const float* __restrict__ x_l1,
                                 const float* __restrict__ score_k, float* __restrict__ x_up, int K) {
    int idx = blockIdx.x * blockDim.x + threadIdx.x;       // over K*64
    if (idx >= K * 64) return;
    int pos = idx >> 6, c = idx & 63;
    x_up[idx] = x_l1[(size_t)permI[pos] * 64 + c] * tanhf(score_k[pos]);
}

// LN+GELU of hinge GEMM, chiral mix, x_l2_down broadcast, col-sum partials
__global__ void chiral_kernel(const float* __restrict__ x_up, const float* __restrict__ t_ul,
                              const float* __restrict__ g_ul, const float* __restrict__ bn_ul,
                              const float* __restrict__ rows, float* __restrict__ x_down_out,
                              float* __restrict__ chiral_out, float* __restrict__ cs_partial, int K) {
    __shared__ float cs[64];
    if (threadIdx.x < 64) cs[threadIdx.x] = 0.f;
    __syncthreads();
    int k = blockIdx.x * blockDim.x + threadIdx.x;         // blockDim = 128
    if (k < K) {
        const float* tr = t_ul + (size_t)k * 64;
        float mu = 0;  for (int c = 0; c < 64; ++c) mu += tr[c];
        mu *= (1.f / 64.f);
        float var = 0; for (int c = 0; c < 64; ++c) { float x = tr[c] - mu; var += x * x; }
        float inv = rsqrtf(var * (1.f / 64.f) + 1e-5f);
        for (int c = 0; c < 64; ++c) {
            float h  = (tr[c] - mu) * inv * g_ul[c] + bn_ul[c];
            float ut = 0.5f * h * (1.f + erff(h * 0.7071067811865475f));
            float up = x_up[(size_t)k * 64 + c];
            float dn = rows[c];
            float a = rows[128 + c], b = rows[192 + c];
            float ch = 0.5f * ((a * up + (1.f - a) * rows[64 + c]) +
                               (b * dn + (1.f - b) * ut));
            chiral_out[(size_t)k * 64 + c] = ch;
            x_down_out[(size_t)k * 64 + c] = dn;
            atomicAdd(&cs[c], ch);                         // LDS ds_add_f32
        }
    }
    __syncthreads();
    if (threadIdx.x < 64) cs_partial[(size_t)blockIdx.x * 64 + threadIdx.x] = cs[threadIdx.x];
}

__global__ void colsum_final_kernel(const float* cs_partial, int nb, float* cs) {
    int c = threadIdx.x;                                   // 64 threads
    float s = 0;
    for (int b = 0; b < nb; ++b) s += cs_partial[(size_t)b * 64 + c];
    cs[c] = s;
}

__global__ void predictor_kernel(const float* cs, const float* w_p1, const float* b_p1,
                                 const float* w_p2, const float* b_p2, float* logits, int K) {
    __shared__ float xg[64], h[32];
    int t = threadIdx.x;                                   // 64 threads
    xg[t] = cs[t] / (float)K;
    __syncthreads();
    if (t < 32) {
        float v = b_p1[t];
        for (int c = 0; c < 64; ++c) v += xg[c] * w_p1[c * 32 + t];
        h[t] = fmaxf(v, 0.f);
    }
    __syncthreads();
    if (t < 16) {
        float v = b_p2[t];
        for (int j = 0; j < 32; ++j) v += h[j] * w_p2[j * 16 + t];
        logits[t] = v;
    }
}

__global__ void cycle_partial_kernel(const float* __restrict__ rec, const float* __restrict__ x_l1,
                                     float* __restrict__ part, size_t n) {
    __shared__ float sd[256];
    float s = 0.f;
    for (size_t i = (size_t)blockIdx.x * blockDim.x + threadIdx.x; i < n;
         i += (size_t)gridDim.x * blockDim.x) {
        float d = rec[i] - x_l1[i];
        s += d * d;
    }
    sd[threadIdx.x] = s;
    __syncthreads();
    for (int st = 128; st > 0; st >>= 1) {
        if (threadIdx.x < st) sd[threadIdx.x] += sd[threadIdx.x + st];
        __syncthreads();
    }
    if (threadIdx.x == 0) part[blockIdx.x] = sd[0];
}

__global__ void cycle_final_kernel(const float* part, int nb, float* out, float scale) {
    if (threadIdx.x == 0) {
        float s = 0;
        for (int i = 0; i < nb; ++i) s += part[i];
        out[0] = s * scale;
    }
}

// ===========================================================================
extern "C" void kernel_launch(void* const* d_in, const int* in_sizes, int n_in,
                              void* d_out, int out_size, void* d_ws, size_t ws_size,
                              hipStream_t stream) {
    const float* x        = (const float*)d_in[0];
    const int*   ei       = (const int*)d_in[1];
    const int*   et       = (const int*)d_in[2];
    const float* basis    = (const float*)d_in[3];
    const float* comp     = (const float*)d_in[4];
    const float* w_root   = (const float*)d_in[5];
    const float* b_rgcn   = (const float*)d_in[6];
    const float* pool_p   = (const float*)d_in[7];
    const float* l3_prior = (const float*)d_in[8];
    const float* w_unpool = (const float*)d_in[9];
    const float* b_unpool = (const float*)d_in[10];
    const float* alpha    = (const float*)d_in[11];
    const float* beta     = (const float*)d_in[12];
    const float* w_lu     = (const float*)d_in[13];
    const float* b_lu     = (const float*)d_in[14];
    const float* g_lu     = (const float*)d_in[15];
    const float* bn_lu    = (const float*)d_in[16];
    const float* w_ul     = (const float*)d_in[17];
    const float* b_ul     = (const float*)d_in[18];
    const float* g_ul     = (const float*)d_in[19];
    const float* bn_ul    = (const float*)d_in[20];
    const float* w_p1     = (const float*)d_in[21];
    const float* b_p1     = (const float*)d_in[22];
    const float* w_p2     = (const float*)d_in[23];
    const float* b_p2     = (const float*)d_in[24];
    const float* w_rec    = (const float*)d_in[25];
    const float* b_rec    = (const float*)d_in[26];

    float*    ws    = (float*)d_ws;
    float*    out   = (float*)d_out;
    unsigned* hist  = (unsigned*)(ws + OFF_HIST);
    unsigned* state = (unsigned*)(ws + OFF_STATE);
    unsigned* keys  = (unsigned*)(ws + OFF_KEYS);
    unsigned* blkG  = (unsigned*)(ws + OFF_BLKG);
    unsigned* blkT  = (unsigned*)(ws + OFF_BLKT);
    unsigned* prefG = (unsigned*)(ws + OFF_PREFG);
    unsigned* prefT = (unsigned*)(ws + OFF_PREFT);
    int*      permI = (int*)(ws + OFF_PERMI);

    // 0) zero agg/deg/hist/state (contiguous) and x_l1_rec output region
    size_t zspan = (size_t)N_NODES * 64 + N_NODES + 256 + 8;
    fill_zero_kernel<<<1024, 256, 0, stream>>>(ws + OFF_AGG, zspan);
    fill_zero_kernel<<<1024, 256, 0, stream>>>(out + O_REC, (size_t)N_NODES * 64);

    // 1) single-row params: down_row, lower_t, sigmoids, ||pool_p||
    rows_kernel<<<1, 64, 0, stream>>>(l3_prior, w_unpool, b_unpool, w_lu, b_lu, g_lu, bn_lu,
                                      alpha, beta, pool_p, ws + OFF_ROWS);

    // 2) fused WMMA triple GEMM: xb0, xb1, x@w_root
    rgcn_gemms_kernel<<<(N_NODES + 127) / 128, 256, 0, stream>>>(
        x, basis, w_root, ws + OFF_XB, ws + OFF_XL1, N_NODES);

    // 3) edge messages: one wave per edge, scatter-add
    edge_msg_kernel<<<N_EDGES / 8, 256, 0, stream>>>(ei, et, comp, ws + OFF_XB,
                                                     ws + OFF_AGG, ws + OFF_DEG, N_EDGES);

    // 4) finalize x_l1 (in place), score, radix keys
    finalize_l1_kernel<<<NBLK_N, 256, 0, stream>>>(ws + OFF_XL1, ws + OFF_AGG, ws + OFF_DEG,
                                                   b_rgcn, pool_p, ws + OFF_ROWS,
                                                   ws + OFF_SCORE, keys, N_NODES);

    // 5) exact top-K threshold: 4-pass MSB radix select
    for (int p = 0; p < 4; ++p) {
        radix_hist_kernel<<<NBLK_N, 256, 0, stream>>>(keys, N_NODES, state, p, hist);
        radix_pick_kernel<<<1, 256, 0, stream>>>(hist, state, p, POOL_K);
    }

    // 6) deterministic compaction -> perm / score_k / x_l2_up
    select_flags_kernel<<<NBLK_N, 256, 0, stream>>>(keys, N_NODES, state, blkG, blkT);
    block_scan_kernel<<<1, 64, 0, stream>>>(blkG, blkT, prefG, prefT, NBLK_N);
    emit_perm_kernel<<<NBLK_N, 256, 0, stream>>>(keys, ws + OFF_SCORE, N_NODES, POOL_K, state,
                                                 prefG, prefT, permI, out + O_PERM, out + O_SCOREK);
    gather_up_kernel<<<(POOL_K * 64 + 255) / 256, 256, 0, stream>>>(
        permI, ws + OFF_XL1, out + O_SCOREK, out + O_UP, POOL_K);

    // 7) hinge GEMM (WMMA + TDM weight staging): t_ul = x_l2_up @ w_ul + b_ul
    gemm64_bias_kernel<<<(POOL_K + 127) / 128, 256, 0, stream>>>(
        out + O_UP, w_ul, b_ul, ws + OFF_TUL, POOL_K);

    // 8) LN+GELU, chiral mix, x_l2_down broadcast, col-sum partials
    chiral_kernel<<<NBLK_K, 128, 0, stream>>>(out + O_UP, ws + OFF_TUL, g_ul, bn_ul,
                                              ws + OFF_ROWS, out + O_DOWN, out + O_CHIRAL,
                                              ws + OFF_CSP, POOL_K);
    colsum_final_kernel<<<1, 64, 0, stream>>>(ws + OFF_CSP, NBLK_K, ws + OFF_CS);

    // 9) predictor MLP -> logits
    predictor_kernel<<<1, 64, 0, stream>>>(ws + OFF_CS, w_p1, b_p1, w_p2, b_p2,
                                           out + O_LOGITS, POOL_K);

    // 10) recon GEMM (WMMA + TDM) + scatter to x_l1_rec[perm]
    recon_scatter_kernel<<<(POOL_K + 127) / 128, 256, 0, stream>>>(
        out + O_CHIRAL, w_rec, b_rec, permI, out + O_REC, POOL_K);

    // 11) cycle loss = mean((x_l1_rec - x_l1)^2)
    cycle_partial_kernel<<<CYC_BLK, 256, 0, stream>>>(out + O_REC, ws + OFF_XL1,
                                                      ws + OFF_CYC, (size_t)N_NODES * 64);
    cycle_final_kernel<<<1, 64, 0, stream>>>(ws + OFF_CYC, CYC_BLK, out + O_CYCLE,
                                             1.0f / (float)((size_t)N_NODES * 64));
}
